// ann1_snn7_69621419868180
// MI455X (gfx1250) — matrix operand order repaired and checked by reading, not compiled, not run on hardware
//
#include <hip/hip_runtime.h>
#include <hip/hip_bf16.h>

// ---------------------------------------------------------------------------
// SNN pipeline for MI455X (gfx1250, wave32, WMMA + TDM).
//   convs / FCs  : implicit-GEMM, v_wmma_f32_16x16x32_f16, 16x32 tile per wave
//                  dense 2D tiles staged via tensor_load_to_lds (TDM) when
//                  available; im2col gather staged manually in f16 LDS
//   IIR / LIF    : [T,N] layout, thread-per-column sequential scan, in place
// ---------------------------------------------------------------------------

typedef __attribute__((ext_vector_type(16))) _Float16     v16h;
typedef __attribute__((ext_vector_type(8)))  float        v8f;
typedef __attribute__((ext_vector_type(4)))  unsigned int v4u;
typedef __attribute__((ext_vector_type(8)))  int          v8i;
typedef __attribute__((ext_vector_type(4)))  int          v4i;

#define DECAY_M 0.7788007830714049f   /* exp(-1/4) */
#define DECAY_S 0.36787944117144233f  /* exp(-1)   */
#define T_STEPS 100
#define BATCH   16
#define NIMG    (T_STEPS * BATCH)

#if __has_builtin(__builtin_amdgcn_tensor_load_to_lds) && \
    __has_builtin(__builtin_amdgcn_s_wait_tensorcnt)
#define USE_TDM 1
#else
#define USE_TDM 0
#endif

// ---------------------------------------------------------------------------
// Tensor Data Mover: DMA a dense 2D f32 tile (tile_w contiguous elems x
// tile_h rows, row stride row_stride elems) from global into LDS, row-major.
// OOB rows/cols (beyond tens_w/tens_h) are zero-filled by the TDM.
// 6-arg builtin form: (g0, g1, g2, g3, g_extra, cpol).
// ---------------------------------------------------------------------------
#if USE_TDM
__device__ inline void tdm_load_2d_f32(const float* gsrc, void* lds_dst,
                                       unsigned int tile_w, unsigned int tile_h,
                                       unsigned long long row_stride,
                                       unsigned int tens_w, unsigned int tens_h)
{
    unsigned long long ga = (unsigned long long)(size_t)gsrc;
    unsigned int laddr    = (unsigned int)(size_t)lds_dst;   // LDS byte offset
    // D# group 0: [1:0]=count=1 | lds_addr[63:32] | global_addr[120:64] | type=2[127:126]
    v4u g0;
    g0[0] = 1u;
    g0[1] = laddr;
    g0[2] = (unsigned int)(ga & 0xffffffffull);
    g0[3] = (unsigned int)((ga >> 32) & 0x01ffffffull) | (2u << 30);
    // D# group 1: data_size[17:16]=2 (4B); tensor_dim0[79:48]; tensor_dim1[111:80];
    //             tile_dim0[127:112]; tile_dim1[143:128]; tensor_dim0_stride[207:160]
    v8i g1;
    g1[0] = (int)(2u << 16);
    g1[1] = (int)((tens_w & 0xffffu) << 16);
    g1[2] = (int)(((tens_w >> 16) & 0xffffu) | ((tens_h & 0xffffu) << 16));
    g1[3] = (int)(((tens_h >> 16) & 0xffffu) | ((tile_w & 0xffffu) << 16));
    g1[4] = (int)(tile_h & 0xffffu);                 // tile_dim1; tile_dim2 = 0
    g1[5] = (int)(row_stride & 0xffffffffull);
    g1[6] = (int)((row_stride >> 32) & 0xffffull);   // stride[47:32]; dim1_stride = 0
    g1[7] = 0;
    v4i z4 = {0, 0, 0, 0};
    v8i z8 = {0, 0, 0, 0, 0, 0, 0, 0};
    __builtin_amdgcn_tensor_load_to_lds(g0, g1, z4, z4, z8, 0);
}
#endif

// ---------------------------------------------------------------------------
// Elementwise / recurrence kernels
// ---------------------------------------------------------------------------

__global__ void sig_iir_kernel(const float* __restrict__ in,
                               const float* __restrict__ a1,
                               const float* __restrict__ a2,
                               float* __restrict__ y, int N, int alphaN)
{
    int n = blockIdx.x * blockDim.x + threadIdx.x;
    if (n >= N) return;
    float c1 = a1[n % alphaN];
    float c2 = a2[n % alphaN];
    const float* xin = in + (size_t)n * T_STEPS;
    float p1 = 0.f, p2 = 0.f;
    for (int t = 0; t < T_STEPS; ++t) {
        float s  = 1.f / (1.f + expf(-xin[t]));
        float yv = c1 * p1 + c2 * p2 + s;
        y[(size_t)t * N + n] = yv;
        p2 = p1; p1 = yv;
    }
}

__global__ void iir_kernel(float* __restrict__ x,
                           const float* __restrict__ a1,
                           const float* __restrict__ a2, int N, int alphaN)
{
    int n = blockIdx.x * blockDim.x + threadIdx.x;
    if (n >= N) return;
    float c1 = a1[n % alphaN];
    float c2 = a2[n % alphaN];
    float p1 = 0.f, p2 = 0.f;
    for (int t = 0; t < T_STEPS; ++t) {
        size_t idx = (size_t)t * N + n;
        if (t + 1 < T_STEPS) __builtin_prefetch(&x[idx + N], 0, 0);
        float yv = c1 * p1 + c2 * p2 + x[idx];
        x[idx] = yv;
        p2 = p1; p1 = yv;
    }
}

__global__ void lif_kernel(float* __restrict__ c, int N)
{
    int n = blockIdx.x * blockDim.x + threadIdx.x;
    if (n >= N) return;
    float v = 0.f, s = 0.f;
    for (int t = 0; t < T_STEPS; ++t) {
        size_t idx = (size_t)t * N + n;
        float vn = v * DECAY_M * (1.f - s) + c[idx];
        s = (vn >= 1.f) ? 1.f : 0.f;
        c[idx] = s;
        v = vn;
    }
}

__global__ void lif_out_kernel(const float* __restrict__ c, float* __restrict__ out)
{
    int n = blockIdx.x * blockDim.x + threadIdx.x;
    if (n >= BATCH * 10) return;
    int b = n / 10, o = n % 10;
    float v = 0.f, s = 0.f;
    for (int t = 0; t < T_STEPS; ++t) {
        float vn = v * DECAY_M * (1.f - s) + c[(size_t)t * (BATCH * 16) + b * 16 + o];
        s = (vn >= 1.f) ? 1.f : 0.f;
        out[(size_t)(b * 10 + o) * T_STEPS + t] = s;
        v = vn;
    }
}

__global__ void maxpool_kernel(const float* __restrict__ in, float* __restrict__ out,
                               int C, int Hi, int Wi, long total)
{
    long i = (long)blockIdx.x * blockDim.x + threadIdx.x;
    if (i >= total) return;
    int Ho = Hi >> 1, Wo = Wi >> 1;
    int wo = (int)(i % Wo); long r = i / Wo;
    int ho = (int)(r % Ho); r /= Ho;
    int cc = (int)(r % C);  long img = r / C;
    const float* p = in + (((img * C + cc) * (long)Hi + 2 * ho) * Wi + 2 * wo);
    out[i] = fmaxf(fmaxf(p[0], p[1]), fmaxf(p[Wi], p[Wi + 1]));
}

// ---------------------------------------------------------------------------
// WMMA fragment helpers (wave32, 16-bit A 16x32 / B 32x16 lane layouts)
// ---------------------------------------------------------------------------

// A fragment from f32 LDS tile [16][32] row-major (cvt f32->f16 at load)
__device__ inline v16h frag_a_f32(const float* A32, int lane)
{
    int hl = lane >> 4, m = lane & 15;
    v16h a;
#pragma unroll
    for (int r = 0; r < 8; ++r) {
        int kb = (hl ? 16 : 0) + 2 * r;
        a[2 * r]     = (_Float16)A32[m * 32 + kb];
        a[2 * r + 1] = (_Float16)A32[m * 32 + kb + 1];
    }
    return a;
}

// B fragment from f16 LDS tile [32][stride] k-major
__device__ inline v16h frag_b_f16(const _Float16* Bs, int lane, int colBase, int stride)
{
    int hl = lane >> 4;
    int nc = colBase + (lane & 15);
    v16h b;
#pragma unroll
    for (int r = 0; r < 8; ++r) {
        int kb = (hl ? 16 : 0) + 2 * r;
        b[2 * r]     = Bs[kb * stride + nc];
        b[2 * r + 1] = Bs[(kb + 1) * stride + nc];
    }
    return b;
}

// B fragment from f32 LDS tile [nrows][32] row-major, where B-column = row index
__device__ inline v16h frag_b_rowmajor_f32(const float* B32, int lane, int colBase)
{
    int hl = lane >> 4;
    int oc = colBase + (lane & 15);
    v16h b;
#pragma unroll
    for (int r = 0; r < 8; ++r) {
        int kb = (hl ? 16 : 0) + 2 * r;
        b[2 * r]     = (_Float16)B32[oc * 32 + kb];
        b[2 * r + 1] = (_Float16)B32[oc * 32 + kb + 1];
    }
    return b;
}

// ---------------------------------------------------------------------------
// Implicit-GEMM 3x3 VALID conv. Block = 128 thr (4 waves):
//   16 output channels x 128 output pixels; each wave owns a 16x32 tile
//   (two accumulators, shared A fragment, 2 WMMAs per K-step).
// ---------------------------------------------------------------------------
#define BSTRIDE 136
__global__ __launch_bounds__(128)
void conv_wmma_kernel(const float* __restrict__ in, const float* __restrict__ w,
                      const float* __restrict__ bias, float* __restrict__ out,
                      int Cin, int H, int W, int Cout, int Ho, int Wo)
{
    const int img      = blockIdx.z;
    const int coutBase = blockIdx.y * 16;
    const int pixBase  = blockIdx.x * 128;
    const int K        = Cin * 9;
    const int HoWo     = Ho * Wo;
    const int tid      = threadIdx.x;
    const int lane     = tid & 31;
    const int wave     = tid >> 5;

    __shared__ float    As32[16 * 32];       // weight tile (TDM target)
    __shared__ _Float16 Bs[32 * BSTRIDE];    // im2col patch, f16

    const float* inImg = in + (size_t)img * Cin * H * W;
    v8f acc0 = {}, acc1 = {};

    for (int k0 = 0; k0 < K; k0 += 32) {
#if USE_TDM
        if (wave == 0)   // weights: dense 2D tile, stride-K rows -> TDM
            tdm_load_2d_f32(&w[(size_t)coutBase * K + k0], As32,
                            32u, 16u, (unsigned long long)K,
                            (unsigned int)(K - k0), 16u);
#else
        for (int i = tid; i < 512; i += 128) {
            int m = i >> 5, k = i & 31;
            As32[i] = w[(size_t)(coutBase + m) * K + k0 + k];
        }
#endif
        // im2col gather 32x128 -> f16 LDS (manual: scatter/gather shaped)
        for (int i = tid; i < 4096; i += 128) {
            int k = i >> 7, p = i & 127;
            int kk = k0 + k, op = pixBase + p;
            float v = 0.f;
            if (op < HoWo) {
                int ci = kk / 9;
                int rr = (kk % 9) / 3;
                int ss = kk % 3;
                int oh = op / Wo, ow = op % Wo;
                v = inImg[(size_t)ci * H * W + (size_t)(oh + rr) * W + (ow + ss)];
            }
            Bs[k * BSTRIDE + p] = (_Float16)v;
        }
#if USE_TDM
        if (wave == 0) __builtin_amdgcn_s_wait_tensorcnt(0);
#endif
        __syncthreads();

        v16h a  = frag_a_f32(As32, lane);
        v16h b0 = frag_b_f16(Bs, lane, wave * 32, BSTRIDE);
        v16h b1 = frag_b_f16(Bs, lane, wave * 32 + 16, BSTRIDE);
        acc0 = __builtin_amdgcn_wmma_f32_16x16x32_f16(false, a, false, b0,
                                                      (short)0, acc0, false, false);
        acc1 = __builtin_amdgcn_wmma_f32_16x16x32_f16(false, a, false, b1,
                                                      (short)0, acc1, false, false);
        __syncthreads();
    }

    // D layout: reg r, lanes 0-15 -> M=r, lanes 16-31 -> M=8+r; N = lane&15
    int ncol  = lane & 15;
    int mhalf = (lane >> 4) * 8;
    size_t obase = (size_t)img * Cout * HoWo;
    int p0 = pixBase + wave * 32 + ncol;
    int p1 = p0 + 16;
#pragma unroll
    for (int r = 0; r < 8; ++r) {
        int m = coutBase + mhalf + r;
        float bm = bias[m];
        if (p0 < HoWo) out[obase + (size_t)m * HoWo + p0] = acc0[r] + bm;
        if (p1 < HoWo) out[obase + (size_t)m * HoWo + p1] = acc1[r] + bm;
    }
}

// ---------------------------------------------------------------------------
// FC GEMM: rows M = (t,b) images, cols N = output features.
// Block = 128 thr: 16 rows x 128 outs; both tiles are dense 2D -> TDM.
// out[row * NoutStride + o] = sum_k x[row*K+k] * w[o*K+k] + bias[o]
// ---------------------------------------------------------------------------
__global__ __launch_bounds__(128)
void fc_wmma_kernel(const float* __restrict__ x, const float* __restrict__ w,
                    const float* __restrict__ bias, float* __restrict__ out,
                    int K, int Nout, int NoutStride)
{
    const int rowBase = blockIdx.x * 16;
    const int outBase = blockIdx.y * 128;
    const int tid     = threadIdx.x;
    const int lane    = tid & 31;
    const int wave    = tid >> 5;

    __shared__ float As32[16 * 32];    // activations tile
    __shared__ float Bw32[128 * 32];   // weight tile, [out][k] row-major

    v8f acc0 = {}, acc1 = {};

    for (int k0 = 0; k0 < K; k0 += 32) {
#if USE_TDM
        if (wave == 0) {
            tdm_load_2d_f32(&x[(size_t)rowBase * K + k0], As32,
                            32u, 16u, (unsigned long long)K,
                            (unsigned int)(K - k0), 16u);
            // rows beyond Nout zero-filled by TDM OOB handling (fc8 padding)
            tdm_load_2d_f32(&w[(size_t)outBase * K + k0], Bw32,
                            32u, 128u, (unsigned long long)K,
                            (unsigned int)(K - k0), (unsigned int)(Nout - outBase));
            __builtin_amdgcn_s_wait_tensorcnt(0);
        }
#else
        for (int i = tid; i < 512; i += 128) {
            int m = i >> 5, k = i & 31;
            As32[i] = x[(size_t)(rowBase + m) * K + k0 + k];
        }
        for (int i = tid; i < 4096; i += 128) {
            int o = i >> 5, k = i & 31;
            int oo = outBase + o;
            Bw32[i] = (oo < Nout) ? w[(size_t)oo * K + k0 + k] : 0.f;
        }
#endif
        __syncthreads();

        v16h a  = frag_a_f32(As32, lane);
        v16h b0 = frag_b_rowmajor_f32(Bw32, lane, wave * 32);
        v16h b1 = frag_b_rowmajor_f32(Bw32, lane, wave * 32 + 16);
        acc0 = __builtin_amdgcn_wmma_f32_16x16x32_f16(false, a, false, b0,
                                                      (short)0, acc0, false, false);
        acc1 = __builtin_amdgcn_wmma_f32_16x16x32_f16(false, a, false, b1,
                                                      (short)0, acc1, false, false);
        __syncthreads();
    }

    int ncol  = lane & 15;
    int mhalf = (lane >> 4) * 8;
    int o0 = outBase + wave * 32 + ncol;
    int o1 = o0 + 16;
#pragma unroll
    for (int r = 0; r < 8; ++r) {
        int row = rowBase + mhalf + r;
        if (o0 < Nout) out[(size_t)row * NoutStride + o0] = acc0[r] + bias[o0];
        if (o1 < Nout) out[(size_t)row * NoutStride + o1] = acc1[r] + bias[o1];
    }
}

// ---------------------------------------------------------------------------
// Host-side pipeline
// ---------------------------------------------------------------------------
extern "C" void kernel_launch(void* const* d_in, const int* in_sizes, int n_in,
                              void* d_out, int out_size, void* d_ws, size_t ws_size,
                              hipStream_t stream)
{
    (void)in_sizes; (void)n_in; (void)out_size; (void)ws_size;

    const float* inp  = (const float*)d_in[0];
    const float* a2_1 = (const float*)d_in[1];  const float* a2_2 = (const float*)d_in[2];
    const float* a3_1 = (const float*)d_in[3];  const float* a3_2 = (const float*)d_in[4];
    const float* a4_1 = (const float*)d_in[5];  const float* a4_2 = (const float*)d_in[6];
    const float* a5_1 = (const float*)d_in[7];  const float* a5_2 = (const float*)d_in[8];
    const float* a6_1 = (const float*)d_in[9];  const float* a6_2 = (const float*)d_in[10];
    const float* a7_1 = (const float*)d_in[11]; const float* a7_2 = (const float*)d_in[12];
    const float* a8_1 = (const float*)d_in[13]; const float* a8_2 = (const float*)d_in[14];
    const float* w2 = (const float*)d_in[15]; const float* b2 = (const float*)d_in[16];
    const float* w3 = (const float*)d_in[17]; const float* b3 = (const float*)d_in[18];
    const float* w5 = (const float*)d_in[19]; const float* b5 = (const float*)d_in[20];
    const float* w7 = (const float*)d_in[21]; const float* b7 = (const float*)d_in[22];
    const float* w8 = (const float*)d_in[23]; const float* b8 = (const float*)d_in[24];
    float* out = (float*)d_out;
    float* ws  = (float*)d_ws;

    const int N1 = BATCH * 32 * 32 * 32;   // 524288
    const int N2 = BATCH * 32 * 30 * 30;   // 460800
    const int N3 = BATCH * 64 * 28 * 28;   // 802816
    const int N4 = BATCH * 64 * 14 * 14;   // 200704
    const int N5 = BATCH * 64 * 12 * 12;   // 147456
    const int N6 = BATCH * 2304;           // 36864
    const int N7 = BATCH * 512;            // 8192

    // lifetime-overlapped workspace (floats); peak ~506 MB
    float* B2 = ws;                        // [T,N2] conv2 out
    float* X1 = ws + 46080000;             // [T,N1] iir2 out (conv2 input)
    float* C3 = ws + 46080000;             // [T,N3] conv3 out (reuses X1)
    float* D4 = ws;                        // [T,N4] pool1 out (reuses B2)
    float* E5 = ws + 20070400;             // [T,N5] conv5 out
    float* F6 = ws + 34816000;             // [T,N6] pool2 out
    float* G7 = ws + 38502400;             // [T,N7] fc7 out
    float* H8 = ws + 39321600;             // [T,B*16] fc8 out

    const int TB = 256;
    #define NB(n) (((n) + TB - 1) / TB)

    sig_iir_kernel<<<NB(N1), TB, 0, stream>>>(inp, a2_1, a2_2, X1, N1, 32 * 32 * 32);

    conv_wmma_kernel<<<dim3(8, 2, NIMG), 128, 0, stream>>>(X1, w2, b2, B2, 32, 32, 32, 32, 30, 30);
    lif_kernel<<<NB(N2), TB, 0, stream>>>(B2, N2);
    iir_kernel<<<NB(N2), TB, 0, stream>>>(B2, a3_1, a3_2, N2, 32 * 30 * 30);

    conv_wmma_kernel<<<dim3(7, 4, NIMG), 128, 0, stream>>>(B2, w3, b3, C3, 32, 30, 30, 64, 28, 28);
    lif_kernel<<<NB(N3), TB, 0, stream>>>(C3, N3);
    iir_kernel<<<NB(N3), TB, 0, stream>>>(C3, a4_1, a4_2, N3, 64 * 28 * 28);

    {
        long total = (long)NIMG * 64 * 14 * 14;
        maxpool_kernel<<<NB(total), TB, 0, stream>>>(C3, D4, 64, 28, 28, total);
    }
    iir_kernel<<<NB(N4), TB, 0, stream>>>(D4, a5_1, a5_2, N4, 64 * 14 * 14);

    conv_wmma_kernel<<<dim3(2, 4, NIMG), 128, 0, stream>>>(D4, w5, b5, E5, 64, 14, 14, 64, 12, 12);
    lif_kernel<<<NB(N5), TB, 0, stream>>>(E5, N5);
    iir_kernel<<<NB(N5), TB, 0, stream>>>(E5, a6_1, a6_2, N5, 64 * 12 * 12);

    {
        long total = (long)NIMG * 64 * 6 * 6;
        maxpool_kernel<<<NB(total), TB, 0, stream>>>(E5, F6, 64, 12, 12, total);
    }
    iir_kernel<<<NB(N6), TB, 0, stream>>>(F6, a7_1, a7_2, N6, 2304);

    fc_wmma_kernel<<<dim3(NIMG / 16, 512 / 128), 128, 0, stream>>>(F6, w7, b7, G7, 2304, 512, 512);
    lif_kernel<<<NB(N7), TB, 0, stream>>>(G7, N7);
    iir_kernel<<<NB(N7), TB, 0, stream>>>(G7, a8_1, a8_2, N7, 512);

    fc_wmma_kernel<<<dim3(NIMG / 16, 1), 128, 0, stream>>>(G7, w8, b8, H8, 512, 10, 16);
    lif_out_kernel<<<1, 256, 0, stream>>>(H8, out);

    #undef NB
}
